// GCN_27144193311514
// MI455X (gfx1250) — matrix operand order repaired and checked by reading, not compiled, not run on hardware
//
#include <hip/hip_runtime.h>

// ---------------------------------------------------------------------------
// 2-layer GCN (N=100000, E=1600000, dim 48) for gfx1250 (MI455X).
//   hs = (x * dinv[row]) @ W           -> f16 WMMA (v_wmma_f32_16x16x32_f16)
//   acc[col] += hs[row]  (edges)       -> global_atomic_add_f32 (L2 resident)
//   h  = act(dinv * acc + b)           -> elementwise float4
// Working set (~39 MB) fits in the 192 MB L2.
// x-tile staging uses GLOBAL_LOAD_ASYNC_TO_LDS_B128 (ASYNCcnt) to stream the
// raw f32 tile into LDS while W is staged, then converts to f16 in LDS.
// ---------------------------------------------------------------------------

typedef __attribute__((ext_vector_type(16))) _Float16 v16h;
typedef __attribute__((ext_vector_type(8)))  _Float16 v8h;
typedef __attribute__((ext_vector_type(8)))  float    v8f;
typedef __attribute__((ext_vector_type(4)))  int      v4i;

#define DIM 48
#define ROWS_PER_BLOCK 128   // 8 waves x 16-row tiles
#define LDS_STRIDE 72        // halfs per row; 144B stride breaks bank periodicity

union V16U { v16h v; v8h h[2]; };

// ---- async global->LDS (b128) helpers, compile-safe fallbacks ------------
#define GCN_USE_ASYNC 1

__device__ __forceinline__ void async_copy_b128(const float* gp, void* lp) {
#if GCN_USE_ASYNC && __has_builtin(__builtin_amdgcn_global_load_async_to_lds_b128)
    // Builtin expects AS1 (global/__device__) int4* source and AS3 (LDS) int4* dest.
    typedef __attribute__((address_space(1))) v4i* g_v4i_p;
    typedef __attribute__((address_space(3))) v4i* l_v4i_p;
    g_v4i_p g1 = (g_v4i_p)(uintptr_t)gp;
    l_v4i_p l3 = (l_v4i_p)(unsigned int)(uintptr_t)lp;  // low 32 bits = LDS offset
    __builtin_amdgcn_global_load_async_to_lds_b128(g1, l3, 0, 0);
#elif GCN_USE_ASYNC
    unsigned int laddr = (unsigned int)(uintptr_t)lp;
    asm volatile("global_load_async_to_lds_b128 %0, %1, off"
                 :: "v"(laddr), "v"(gp) : "memory");
#else
    *(float4*)lp = *(const float4*)gp;
#endif
}

__device__ __forceinline__ void async_copy_wait0() {
#if GCN_USE_ASYNC
#if __has_builtin(__builtin_amdgcn_s_wait_asynccnt)
    __builtin_amdgcn_s_wait_asynccnt(0);
#else
    asm volatile("s_wait_asynccnt 0x0" ::: "memory");
#endif
#endif
}

// ---------------- degree / dinv ----------------
__global__ void k_fill1(float* __restrict__ p, int n) {
    int i = blockIdx.x * blockDim.x + threadIdx.x;
    if (i < n) p[i] = 1.0f;                       // self-loop contributes 1
}

__global__ void k_deg(const int* __restrict__ col, float* __restrict__ deg, int E) {
    int e = blockIdx.x * blockDim.x + threadIdx.x;
    if (e < E) atomicAdd(&deg[col[e]], 1.0f);
}

__global__ void k_rsqrt_inplace(float* __restrict__ p, int n) {
    int i = blockIdx.x * blockDim.x + threadIdx.x;
    if (i < n) p[i] = rsqrtf(p[i]);               // deg >= 1 always
}

// ---------------- GEMM: hs = (in * dinv[row]) @ W, via WMMA ----------------
// Writes hs to outHs AND outAcc (acc initialized with self-loop term).
// outAcc may alias `in`: each block reads only its own 128 rows into LDS
// (before __syncthreads) and writes only those same rows afterwards.
__global__ __launch_bounds__(256)
void k_gemm48_wmma(const float* __restrict__ in, const float* __restrict__ W,
                   const float* __restrict__ dinv,
                   float* __restrict__ outHs, float* __restrict__ outAcc, int n)
{
    __shared__ __align__(16) float    sX[ROWS_PER_BLOCK * DIM];        // raw f32 tile
    __shared__ __align__(16) _Float16 sA[ROWS_PER_BLOCK * LDS_STRIDE]; // [row][k] f16
    __shared__ __align__(16) _Float16 sB[DIM * LDS_STRIDE];            // [nOut][k] (W^T)

    const int tid  = threadIdx.x;
    const int base = blockIdx.x * ROWS_PER_BLOCK;

    // 1) Kick off async f32 tile copy: global -> LDS, 16B per issue.
    for (int f = tid; f < ROWS_PER_BLOCK * 12; f += 256) {
        int r = f / 12, c4 = f % 12;
        int g = base + r; if (g >= n) g = n - 1;              // clamp (stores masked)
        async_copy_b128(in + (size_t)g * DIM + c4 * 4, &sX[r * DIM + c4 * 4]);
    }

    // 2) Stage W^T (f16, K zero-padded) while the tile streams in.
    for (int f = tid; f < DIM * 64; f += 256) {
        int nO = f / 64, k = f % 64;
        float w = (k < DIM) ? W[k * DIM + nO] : 0.0f;
        sB[nO * LDS_STRIDE + k] = (_Float16)w;
    }
    // K zero-pad 48..63 of sA
    for (int f = tid; f < ROWS_PER_BLOCK * 16; f += 256) {
        int r = f / 16, k = 48 + (f % 16);
        sA[r * LDS_STRIDE + k] = (_Float16)0.0f;
    }

    async_copy_wait0();
    __syncthreads();

    // 3) Convert f32 tile -> f16, pre-scaled by dinv[row].
    for (int f = tid; f < ROWS_PER_BLOCK * 12; f += 256) {
        int r = f / 12, c4 = f % 12;
        int g = base + r; if (g >= n) g = n - 1;
        float  s  = dinv[g];
        float4 x4 = ((const float4*)sX)[f];
        int o = r * LDS_STRIDE + c4 * 4;
        sA[o + 0] = (_Float16)(x4.x * s);
        sA[o + 1] = (_Float16)(x4.y * s);
        sA[o + 2] = (_Float16)(x4.z * s);
        sA[o + 3] = (_Float16)(x4.w * s);
    }
    __syncthreads();

    const int wave = tid >> 5;       // wave32
    const int lane = tid & 31;
    const int lm   = lane & 15;
    const int hi   = lane >> 4;      // 0: lanes 0-15, 1: lanes 16-31
    const int m0   = base + wave * 16;

    v8f acc0 = {}, acc1 = {}, acc2 = {};

    #pragma unroll
    for (int c = 0; c < 2; ++c) {
        // A fragment (16x32 f16): lane holds row lm; 16 contiguous K from
        // 32c + 16hi  (matches B)
        V16U A;
        const _Float16* ap = sA + (wave * 16 + lm) * LDS_STRIDE + (c * 32 + hi * 16);
        A.h[0] = *(const v8h*)(ap);
        A.h[1] = *(const v8h*)(ap + 8);

        // B fragment (32x16 f16): lane holds col lm; 16 contiguous K from
        // 32c + 16hi  (same k order as A)
        const int kb = c * 32 + hi * 16;
        {
            V16U B; const _Float16* bp = sB + (0 * 16 + lm) * LDS_STRIDE + kb;
            B.h[0] = *(const v8h*)bp;  B.h[1] = *(const v8h*)(bp + 8);
            acc0 = __builtin_amdgcn_wmma_f32_16x16x32_f16(false, A.v, false, B.v,
                                                          (short)0, acc0, false, false);
        }
        {
            V16U B; const _Float16* bp = sB + (1 * 16 + lm) * LDS_STRIDE + kb;
            B.h[0] = *(const v8h*)bp;  B.h[1] = *(const v8h*)(bp + 8);
            acc1 = __builtin_amdgcn_wmma_f32_16x16x32_f16(false, A.v, false, B.v,
                                                          (short)0, acc1, false, false);
        }
        {
            V16U B; const _Float16* bp = sB + (2 * 16 + lm) * LDS_STRIDE + kb;
            B.h[0] = *(const v8h*)bp;  B.h[1] = *(const v8h*)(bp + 8);
            acc2 = __builtin_amdgcn_wmma_f32_16x16x32_f16(false, A.v, false, B.v,
                                                          (short)0, acc2, false, false);
        }
    }

    // D layout: VGPR r -> row m0 + r + 8*hi, col = lm (+16 per N-tile).
    if (base + ROWS_PER_BLOCK <= n) {
        // fast path: whole block in range (uniform scalar branch)
        size_t ro = (size_t)(m0 + hi * 8) * DIM + lm;
        #pragma unroll
        for (int r = 0; r < 8; ++r) {
            outHs[ro]       = acc0[r];
            outHs[ro + 16]  = acc1[r];
            outHs[ro + 32]  = acc2[r];
            outAcc[ro]      = acc0[r];
            outAcc[ro + 16] = acc1[r];
            outAcc[ro + 32] = acc2[r];
            ro += DIM;
        }
    } else {
        #pragma unroll
        for (int r = 0; r < 8; ++r) {
            int row = m0 + hi * 8 + r;
            if (row < n) {
                size_t ro = (size_t)row * DIM;
                outHs[ro + lm]       = acc0[r];
                outHs[ro + 16 + lm]  = acc1[r];
                outHs[ro + 32 + lm]  = acc2[r];
                outAcc[ro + lm]      = acc0[r];
                outAcc[ro + 16 + lm] = acc1[r];
                outAcc[ro + 32 + lm] = acc2[r];
            }
        }
    }
}

// ---------------- edge scatter: acc[col] += hs[row] ----------------
// 12 float4 chunks per edge; 12 consecutive threads share one edge's indices
// (same cacheline). Atomics land in L2 (buffers are L2-resident).
__global__ __launch_bounds__(256)
void k_scatter(const float4* __restrict__ hs4, const int* __restrict__ rowp,
               const int* __restrict__ colp, float* __restrict__ acc, long long total)
{
    long long idx = (long long)blockIdx.x * blockDim.x + threadIdx.x;
    if (idx >= total) return;
    int e = (int)(idx / 12), c = (int)(idx % 12);
    int r = rowp[e], d = colp[e];
    float4 v = hs4[(size_t)r * 12 + c];
    float* p = acc + (size_t)d * DIM + c * 4;
    atomicAdd(p + 0, v.x);
    atomicAdd(p + 1, v.y);
    atomicAdd(p + 2, v.z);
    atomicAdd(p + 3, v.w);
}

// ---------------- finalize: h = act(dinv*acc + b), in place ----------------
__global__ void k_finalize(float* __restrict__ h, const float* __restrict__ dinv,
                           const float* __restrict__ bias, int n, int leaky)
{
    int idx = blockIdx.x * blockDim.x + threadIdx.x;
    if (idx >= n * 12) return;
    int i = idx / 12, c = idx % 12;
    float s = dinv[i];
    float4 v = ((float4*)h)[idx];
    v.x = s * v.x + bias[c * 4 + 0];
    v.y = s * v.y + bias[c * 4 + 1];
    v.z = s * v.z + bias[c * 4 + 2];
    v.w = s * v.w + bias[c * 4 + 3];
    if (leaky) {
        v.x = v.x > 0.f ? v.x : 0.01f * v.x;
        v.y = v.y > 0.f ? v.y : 0.01f * v.y;
        v.z = v.z > 0.f ? v.z : 0.01f * v.z;
        v.w = v.w > 0.f ? v.w : 0.01f * v.w;
    }
    ((float4*)h)[idx] = v;
}

// ---------------- pooling + head ----------------
__global__ void k_zero(float* __restrict__ p, int n) {
    int i = blockIdx.x * blockDim.x + threadIdx.x;
    if (i < n) p[i] = 0.0f;
}

__global__ __launch_bounds__(192)
void k_pool(const float* __restrict__ h, float* __restrict__ gsum, int n)
{
    int k = threadIdx.x % DIM;       // feature
    int g = threadIdx.x / DIM;       // 0..3 row groups per block
    float s = 0.0f;
    for (int i = blockIdx.x * 4 + g; i < n; i += gridDim.x * 4)
        s += h[(size_t)i * DIM + k];
    atomicAdd(&gsum[k], s);
}

__global__ void k_head(const float* __restrict__ gsum, const float* __restrict__ Wlin,
                       const float* __restrict__ blin, float* __restrict__ out)
{
    int j = threadIdx.x;
    if (j < 2) {
        float s = blin[j];
        #pragma unroll
        for (int k = 0; k < DIM; ++k) s += gsum[k] * Wlin[k * 2 + j];
        out[j] = s;
    }
}

// ---------------------------------------------------------------------------
extern "C" void kernel_launch(void* const* d_in, const int* in_sizes, int n_in,
                              void* d_out, int out_size, void* d_ws, size_t ws_size,
                              hipStream_t stream)
{
    const float* x    = (const float*)d_in[0];
    const int*   ei   = (const int*)  d_in[1];   // edge_index [2][E]
    const float* W1   = (const float*)d_in[2];
    const float* b1   = (const float*)d_in[3];
    const float* W2   = (const float*)d_in[4];
    const float* b2   = (const float*)d_in[5];
    const float* Wlin = (const float*)d_in[6];
    const float* blin = (const float*)d_in[7];
    float* out = (float*)d_out;

    const int N = in_sizes[0] / DIM;
    const int E = in_sizes[1] / 2;
    const int* rowp = ei;        // sources
    const int* colp = ei + E;    // targets

    // workspace carve-up (256B aligned)
    char* ws = (char*)d_ws;
    size_t off = 0;
    auto carve = [&](size_t bytes) -> void* {
        void* p = ws + off;
        off = (off + bytes + 255) & ~(size_t)255;
        return p;
    };
    float* dinv = (float*)carve((size_t)N * sizeof(float));
    float* bufA = (float*)carve((size_t)N * DIM * sizeof(float));  // hs
    float* bufB = (float*)carve((size_t)N * DIM * sizeof(float));  // acc / h
    float* gsum = (float*)carve(DIM * sizeof(float));
    (void)ws_size; (void)n_in; (void)out_size;

    const int gN    = (N + 255) / 256;
    const int gE    = (E + 255) / 256;
    const int gGemm = (N + ROWS_PER_BLOCK - 1) / ROWS_PER_BLOCK;
    const long long scT = (long long)E * 12;
    const int gScat = (int)((scT + 255) / 256);
    const int gFin  = (N * 12 + 255) / 256;

    // degree -> dinv = rsqrt(1 + indegree)
    k_fill1        <<<gN, 256, 0, stream>>>(dinv, N);
    k_deg          <<<gE, 256, 0, stream>>>(colp, dinv, E);
    k_rsqrt_inplace<<<gN, 256, 0, stream>>>(dinv, N);

    // ---- layer 1 ----
    k_gemm48_wmma<<<gGemm, 256, 0, stream>>>(x, W1, dinv, bufA, bufB, N);
    k_scatter    <<<gScat, 256, 0, stream>>>((const float4*)bufA, rowp, colp, bufB, scT);
    k_finalize   <<<gFin, 256, 0, stream>>>(bufB, dinv, b1, N, /*leaky=*/1);

    // ---- layer 2 (acc-init aliases the input buffer; per-block safe) ----
    k_gemm48_wmma<<<gGemm, 256, 0, stream>>>(bufB, W2, dinv, bufA, bufB, N);
    k_scatter    <<<gScat, 256, 0, stream>>>((const float4*)bufA, rowp, colp, bufB, scT);
    k_finalize   <<<gFin, 256, 0, stream>>>(bufB, dinv, b2, N, /*leaky=*/0);

    // ---- global add pool + linear head ----
    k_zero<<<1, 64, 0, stream>>>(gsum, DIM);
    k_pool<<<250, 192, 0, stream>>>(bufB, gsum, N);
    k_head<<<1, 64, 0, stream>>>(gsum, Wlin, blin, out);
}
